// CausalBCNAttention_82360292868505
// MI455X (gfx1250) — matrix operand order repaired and checked by reading, not compiled, not run on hardware
//
#include <hip/hip_runtime.h>

typedef __bf16 bf16_t;
typedef __attribute__((ext_vector_type(16))) __bf16 v16bf;
typedef __attribute__((ext_vector_type(8)))  __bf16 v8bf;
typedef __attribute__((ext_vector_type(4)))  __bf16 v4bf;
typedef __attribute__((ext_vector_type(8)))  float  v8f;

#define EPI_STORE_BF16 0
#define EPI_MASK_F32   1
#define EPI_AS_BF16    2
#define EPI_FINAL      3

static __device__ __forceinline__ v16bf make_frag(const bf16_t* p0, const bf16_t* p1){
  union { v16bf v; v8bf h[2]; } u;
  u.h[0] = *(const v8bf*)p0;
  u.h[1] = *(const v8bf*)p1;
  return u.v;
}

static __device__ __forceinline__ void async_b128(unsigned lds_off, unsigned long long gaddr){
  // GLOBAL_LOAD_ASYNC_TO_LDS_B128 (GV mode): LDS[vdst] = MEM[vaddr64], ASYNCcnt-tracked
  asm volatile("global_load_async_to_lds_b128 %0, %1, off"
               :: "v"(lds_off), "v"(gaddr) : "memory");
}
static __device__ __forceinline__ void wait_async0(){
  asm volatile("s_wait_asynccnt 0x0" ::: "memory");
}

// ---------------------------------------------------------------------------
// Main-pass GEMM: all-bf16 operands, async global->LDS staging, double buffer.
// Block tile 256(M) x 128(N), KC=32. 8 waves, wave grid 4(m) x 2(n), each wave
// owns 64x64 = 4x4 subtiles -> 16 v_wmma_f32_16x16x32_bf16 per K-chunk, with
// 4x reuse of both A and B fragments (16 ds_load_b128 per 16 WMMA).
// ---------------------------------------------------------------------------
template<int EPI>
__global__ __launch_bounds__(256) void gemm_async(
    const bf16_t* __restrict__ Ab, const bf16_t* __restrict__ Bb,
    int K, int ldA, int ldB, long strideA,
    void* __restrict__ outb, int ldOut, long strideOut,
    const void* __restrict__ e0b, long strideE0,
    const float* __restrict__ e1b, long strideE1)
{
  constexpr int BM=256, BN=128, KC=32, LSTR=40;     // 40 bf16 = 80B pitch (16B aligned)
  constexpr int ABUF = BM*LSTR, BBUF = BN*LSTR;     // elements
  __shared__ __align__(16) bf16_t sA[2][ABUF];
  __shared__ __align__(16) bf16_t sB[2][BBUF];
  __shared__ float ldsE[BN];

  const int tid  = threadIdx.x;
  const int lane = tid & 31, half = lane >> 4, l16 = lane & 15;
  const int wid  = tid >> 5, wm = wid >> 1, wn = wid & 1;   // 4 x 2 waves
  const int m0 = blockIdx.x * BM, n0 = blockIdx.y * BN;
  const int b  = blockIdx.z;

  const bf16_t* __restrict__ Ag = Ab + (size_t)b * (size_t)strideA;

  if constexpr (EPI==EPI_AS_BF16){
    if (tid < BN){
      // chunk index == blockIdx.x (BM == scan chunk == 256)
      const float* offp = e1b + (size_t)b*(size_t)strideE1 + (size_t)blockIdx.x*(size_t)ldOut;
      ldsE[tid] = offp[n0 + tid];
    }
  } else if constexpr (EPI==EPI_FINAL){
    if (tid < BN) ldsE[tid] = e1b[n0 + tid];
  }

  // ---- async staging addresses: 4 A slots + 2 B slots of 16B per thread/chunk
  unsigned long long gAaddr[4], gBaddr[2];
  unsigned lAoff[4], lBoff[2];
  {
    const int row = tid >> 2, c8 = (tid & 3) * 8;
    #pragma unroll
    for (int j=0;j<4;j++){
      const int r = row + 64*j;                    // rows 0..255
      gAaddr[j] = (unsigned long long)(uintptr_t)(Ag + (size_t)(m0+r)*ldA + c8);
      lAoff[j]  = (unsigned)(uintptr_t)(&sA[0][r*LSTR + c8]);
    }
    #pragma unroll
    for (int j=0;j<2;j++){
      const int r = row + 64*j;                    // rows 0..127
      gBaddr[j] = (unsigned long long)(uintptr_t)(Bb + (size_t)(n0+r)*ldB + c8);
      lBoff[j]  = (unsigned)(uintptr_t)(&sB[0][r*LSTR + c8]);
    }
  }
  constexpr unsigned ABYTES = ABUF*2, BBYTES = BBUF*2;

  auto issue = [&](int bufsel){
    const unsigned at = bufsel ? ABYTES : 0u;
    const unsigned bt = bufsel ? BBYTES : 0u;
    #pragma unroll
    for (int j=0;j<4;j++){ async_b128(lAoff[j]+at, gAaddr[j]); gAaddr[j] += KC*2; }
    #pragma unroll
    for (int j=0;j<2;j++){ async_b128(lBoff[j]+bt, gBaddr[j]); gBaddr[j] += KC*2; }
  };

  v8f acc[4][4];
  #pragma unroll
  for (int im=0;im<4;im++)
    #pragma unroll
    for (int jn=0;jn<4;jn++)
      #pragma unroll
      for (int q=0;q<8;q++) acc[im][jn][q]=0.f;

  const int NCHUNK = K / KC;
  issue(0);
  wait_async0();
  __syncthreads();

  int cur = 0;
  for (int c=0; c<NCHUNK; ++c){
    if (c+1 < NCHUNK) issue(cur^1);     // overlap next chunk's DMA with compute
    const bf16_t* bufA = sA[cur];
    const bf16_t* bufB = sB[cur];
    v16bf bfrag[4];
    #pragma unroll
    for (int jn=0;jn<4;jn++){
      // B 32x16: lanes 0-15 K0..15 (n=lane), lanes 16-31 K16..31
      const bf16_t* bp = bufB + (wn*64 + jn*16 + l16)*LSTR + half*16;
      bfrag[jn] = make_frag(bp, bp+8);
    }
    #pragma unroll
    for (int im=0;im<4;im++){
      // A 16x32: lanes 0-15 K0..15 (m=lane), lanes 16-31 K16..31
      const bf16_t* ap = bufA + (wm*64 + im*16 + l16)*LSTR + half*16;
      v16bf afrag = make_frag(ap, ap+8);
      #pragma unroll
      for (int jn=0;jn<4;jn++)
        acc[im][jn] = __builtin_amdgcn_wmma_f32_16x16x32_bf16(
            false, afrag, false, bfrag[jn], (short)0, acc[im][jn], false, false);
    }
    wait_async0();
    __syncthreads();
    cur ^= 1;
  }

  // ---- epilogue (C layout: VGPR v -> M=v (+8 upper half-wave), N=lane&15)
  #pragma unroll
  for (int im=0;im<4;im++){
    #pragma unroll
    for (int jn=0;jn<4;jn++){
      const int rsub = wm*64 + im*16 + half*8;
      const int col  = n0 + wn*64 + jn*16 + l16;
      #pragma unroll
      for (int v=0; v<8; ++v){
        const int row = m0 + rsub + v;
        const float a = acc[im][jn][v];
        if constexpr (EPI==EPI_MASK_F32){
          const unsigned char* mk = (const unsigned char*)e0b + (size_t)b*(size_t)strideE0;
          float* op = (float*)outb + (size_t)b*(size_t)strideOut;
          op[(size_t)row*ldOut + col] = mk[row] ? a : 0.f;
        } else if constexpr (EPI==EPI_AS_BF16){
          const float* Sp = (const float*)e0b + (size_t)b*(size_t)strideE0;
          bf16_t* op = (bf16_t*)outb + (size_t)b*(size_t)strideOut;
          const float s = Sp[(size_t)row*ldOut + col] + ldsE[col - n0];
          op[(size_t)row*ldOut + col] = (bf16_t)(a * s);
        } else { // EPI_FINAL
          const float* np = (const float*)e0b + (size_t)b*(size_t)strideE0;
          float* op = (float*)outb + (size_t)b*(size_t)strideOut;
          const float nv  = np[row];
          const float rcp = 1.f / fmaxf(nv, 1.f);
          const float ct  = (nv > 0.f) ? ldsE[col - n0] : 0.f;
          op[(size_t)row*ldOut + col] = a * rcp + ct;
        }
      }
    }
  }
}

// ---------------------------------------------------------------------------
// Precompute GEMM (fp32 sources, transpose-capable staging). 128x128 tile,
// KC=64, 8 waves of 64x32 (4x2 subtiles). Same fragment layouts as above.
// ---------------------------------------------------------------------------
template<typename AT, typename BT, int LAYA, int LAYB, int EPI>
__global__ __launch_bounds__(256) void gemm_kernel(
    const AT* __restrict__ Ab, const BT* __restrict__ Bb,
    int K, int ldA, int ldB, long strideA,
    void* __restrict__ outb, int ldOut, long strideOut,
    const void* __restrict__ e0b, long strideE0,
    const float* __restrict__ e1b, long strideE1)
{
  constexpr int BM=128, BN=128, KC=64, LSTR=72;
  __shared__ __align__(16) bf16_t ldsA[BM*LSTR];
  __shared__ __align__(16) bf16_t ldsB[BN*LSTR];

  const int tid  = threadIdx.x;
  const int lane = tid & 31, half = lane >> 4, l16 = lane & 15;
  const int wid  = tid >> 5, wm = wid >> 2, wn = wid & 3;
  const int m0 = blockIdx.x * BM, n0 = blockIdx.y * BN;
  const int b  = blockIdx.z;

  const AT* __restrict__ Ag = Ab + (size_t)b * (size_t)strideA;

  v8f acc[4][2];
  #pragma unroll
  for (int im=0;im<4;im++)
    #pragma unroll
    for (int jn=0;jn<2;jn++)
      #pragma unroll
      for (int q=0;q<8;q++) acc[im][jn][q]=0.f;

  for (int kc=0; kc<K; kc+=KC){
    __syncthreads();
    if constexpr (LAYA==0){
      #pragma unroll
      for (int i=tid; i<BM*(KC/4); i+=256){
        int row=i>>4, c4=(i&15)<<2;
        const float4 vv = *(const float4*)(Ag + (size_t)(m0+row)*ldA + kc + c4);
        v4bf h; h[0]=(bf16_t)vv.x; h[1]=(bf16_t)vv.y; h[2]=(bf16_t)vv.z; h[3]=(bf16_t)vv.w;
        *(v4bf*)(ldsA + row*LSTR + c4) = h;
      }
    } else {
      #pragma unroll
      for (int i=tid; i<KC*(BM/4); i+=256){
        int kr=i>>5, m4=(i&31)<<2;
        const float4 vv = *(const float4*)(Ag + (size_t)(kc+kr)*ldA + m0 + m4);
        ldsA[(m4+0)*LSTR+kr]=(bf16_t)vv.x;
        ldsA[(m4+1)*LSTR+kr]=(bf16_t)vv.y;
        ldsA[(m4+2)*LSTR+kr]=(bf16_t)vv.z;
        ldsA[(m4+3)*LSTR+kr]=(bf16_t)vv.w;
      }
    }
    if constexpr (LAYB==0){
      #pragma unroll
      for (int i=tid; i<BN*(KC/4); i+=256){
        int row=i>>4, c4=(i&15)<<2;
        const float4 vv = *(const float4*)(Bb + (size_t)(n0+row)*ldB + kc + c4);
        v4bf h; h[0]=(bf16_t)vv.x; h[1]=(bf16_t)vv.y; h[2]=(bf16_t)vv.z; h[3]=(bf16_t)vv.w;
        *(v4bf*)(ldsB + row*LSTR + c4) = h;
      }
    } else {
      #pragma unroll
      for (int i=tid; i<KC*(BN/4); i+=256){
        int kr=i>>5, n4=(i&31)<<2;
        const float4 vv = *(const float4*)(Bb + (size_t)(kc+kr)*ldB + n0 + n4);
        ldsB[(n4+0)*LSTR+kr]=(bf16_t)vv.x;
        ldsB[(n4+1)*LSTR+kr]=(bf16_t)vv.y;
        ldsB[(n4+2)*LSTR+kr]=(bf16_t)vv.z;
        ldsB[(n4+3)*LSTR+kr]=(bf16_t)vv.w;
      }
    }
    __syncthreads();
    #pragma unroll
    for (int s=0;s<2;s++){
      const int kb = s*32;
      v16bf bfrag[2];
      #pragma unroll
      for (int jn=0;jn<2;jn++){
        const bf16_t* bp = ldsB + (wn*32 + jn*16 + l16)*LSTR + kb + half*16;
        bfrag[jn] = make_frag(bp, bp+8);
      }
      #pragma unroll
      for (int im=0;im<4;im++){
        const bf16_t* ap = ldsA + (wm*64 + im*16 + l16)*LSTR + kb + half*16;
        v16bf afrag = make_frag(ap, ap+8);
        #pragma unroll
        for (int jn=0;jn<2;jn++)
          acc[im][jn] = __builtin_amdgcn_wmma_f32_16x16x32_bf16(
              false, afrag, false, bfrag[jn], (short)0, acc[im][jn], false, false);
      }
    }
  }

  #pragma unroll
  for (int im=0;im<4;im++){
    #pragma unroll
    for (int jn=0;jn<2;jn++){
      const int rsub = wm*64 + im*16 + half*8;
      const int col  = n0 + wn*32 + jn*16 + l16;
      #pragma unroll
      for (int v=0; v<8; ++v){
        const int row = m0 + rsub + v;
        const float a = acc[im][jn][v];
        if constexpr (EPI==EPI_STORE_BF16){
          ((bf16_t*)outb)[(size_t)row*ldOut + col] = (bf16_t)a;
        }
      }
    }
  }
  (void)e0b; (void)strideE0; (void)e1b; (void)strideE1; (void)strideOut; (void)b;
}

// x (fp32) -> bf16, vectorized
__global__ __launch_bounds__(256) void convert_bf16(const float* __restrict__ in,
                                                    bf16_t* __restrict__ out, long n4)
{
  long i = (long)blockIdx.x*256 + threadIdx.x;
  if (i < n4){
    const float4 v = ((const float4*)in)[i];
    v4bf h; h[0]=(bf16_t)v.x; h[1]=(bf16_t)v.y; h[2]=(bf16_t)v.z; h[3]=(bf16_t)v.w;
    ((v4bf*)out)[i] = h;
  }
}

// Chunked causal prefix sum of Kw over t, in place; chunk totals out.
__global__ __launch_bounds__(256) void scan_chunks(float* __restrict__ Kw,
                                                   float* __restrict__ totals,
                                                   int T, int R, int CH)
{
  const int r = blockIdx.x*256 + threadIdx.x;
  const int chunk = blockIdx.y, b = blockIdx.z;
  size_t base = ((size_t)b*T + (size_t)chunk*CH)*R + r;
  float acc = 0.f;
  for (int t=0; t<CH; ++t){
    acc += Kw[base + (size_t)t*R];
    Kw[base + (size_t)t*R] = acc;
  }
  totals[((size_t)b*gridDim.y + chunk)*R + r] = acc;
}

__global__ __launch_bounds__(256) void scan_totals(float* __restrict__ totals,
                                                   int NCH, int R)
{
  const int r = blockIdx.x*256 + threadIdx.x;
  const int b = blockIdx.z;
  float run = 0.f;
  for (int c=0; c<NCH; ++c){
    size_t i = ((size_t)b*NCH + c)*R + r;
    float v = totals[i];
    totals[i] = run;
    run += v;
  }
}

__global__ __launch_bounds__(256) void mask_scan(const unsigned char* __restrict__ mask,
                                                 float* __restrict__ nvec, int T)
{
  __shared__ float s[256];
  const int b = blockIdx.x, tid = threadIdx.x;
  const int seg = T/256;
  const unsigned char* mp = mask + (size_t)b*T + (size_t)tid*seg;
  float loc = 0.f;
  for (int i=0;i<seg;i++) loc += (mp[i] ? 1.f : 0.f);
  s[tid] = loc; __syncthreads();
  for (int off=1; off<256; off<<=1){
    float v = (tid>=off) ? s[tid-off] : 0.f;
    __syncthreads();
    s[tid] += v;
    __syncthreads();
  }
  float run = (tid>0) ? s[tid-1] : 0.f;
  float* np = nvec + (size_t)b*T + (size_t)tid*seg;
  for (int i=0;i<seg;i++){ run += (mp[i] ? 1.f : 0.f); np[i] = run; }
}

__global__ __launch_bounds__(256) void compute_c(const float* __restrict__ W_O,
                                                 const float* __restrict__ bias,
                                                 float* __restrict__ cvec, int D)
{
  const int e = blockIdx.x*256 + threadIdx.x;
  if (e < D){
    float s = 0.f;
    for (int d=0; d<D; ++d) s += W_O[(size_t)e*D + d] * bias[d];
    cvec[e] = s;
  }
}

extern "C" void kernel_launch(void* const* d_in, const int* in_sizes, int n_in,
                              void* d_out, int out_size, void* d_ws, size_t ws_size,
                              hipStream_t stream)
{
  const float*         x    = (const float*)d_in[0];
  const unsigned char* mask = (const unsigned char*)d_in[1];
  const float*         W_Q  = (const float*)d_in[2];
  const float*         W_K  = (const float*)d_in[3];
  const float*         U    = (const float*)d_in[4];
  const float*         V    = (const float*)d_in[5];
  const float*         Wb   = (const float*)d_in[6];
  const float*         bias = (const float*)d_in[7];
  const float*         W_O  = (const float*)d_in[8];
  (void)in_sizes; (void)n_in; (void)out_size; (void)ws_size;

  constexpr int B=8, T=4096, D=1024, R=1024;
  constexpr int NCH=16, CH=T/NCH;   // CH must equal async-kernel BM (256)

  char* w = (char*)d_ws;
  auto take = [&](size_t bytes)->char*{
    char* p = w; w += (bytes + 255) & ~(size_t)255; return p;
  };
  bf16_t* MqvT = (bf16_t*)take((size_t)R*D*2);
  bf16_t* MkwT = (bf16_t*)take((size_t)R*D*2);
  bf16_t* N2   = (bf16_t*)take((size_t)D*R*2);
  float*  cvec = (float*)take((size_t)D*4);
  float*  nvec = (float*)take((size_t)B*T*4);
  float*  Stot = (float*)take((size_t)B*NCH*R*4);
  bf16_t* xbf  = (bf16_t*)take((size_t)B*T*D*2);
  float*  Kw   = (float*)take((size_t)B*T*R*4);
  bf16_t* AS   = (bf16_t*)take((size_t)B*T*R*2);

  // x -> bf16 once (removes cvt from hot loops, halves A-operand HBM reads)
  convert_bf16<<<dim3((int)(((long)B*T*D/4 + 255)/256)), 256, 0, stream>>>(
      x, xbf, (long)B*T*D/4);

  compute_c<<<dim3((D+255)/256), 256, 0, stream>>>(W_O, bias, cvec, D);
  mask_scan<<<dim3(B), 256, 0, stream>>>(mask, nvec, T);

  // weight-product precomputes (WMMA, bf16 out)
  gemm_kernel<float,float,1,1,EPI_STORE_BF16><<<dim3(R/128, D/128, 1), 256, 0, stream>>>(
      V, W_Q, D, R, D, 0, MqvT, D, 0, nullptr, 0, nullptr, 0);
  gemm_kernel<float,float,1,1,EPI_STORE_BF16><<<dim3(R/128, D/128, 1), 256, 0, stream>>>(
      Wb, W_K, D, R, D, 0, MkwT, D, 0, nullptr, 0, nullptr, 0);
  gemm_kernel<float,float,0,1,EPI_STORE_BF16><<<dim3(D/128, R/128, 1), 256, 0, stream>>>(
      W_O, U, D, D, R, 0, N2, R, 0, nullptr, 0, nullptr, 0);

  // pass 1: Kw[b,t,r] = (x @ MkwT^T) * mask
  gemm_async<EPI_MASK_F32><<<dim3(T/256, R/128, B), 256, 0, stream>>>(
      xbf, MkwT, D, D, D, (long)T*D,
      Kw, R, (long)T*R,
      mask, (long)T, nullptr, 0);

  // pass 2: chunked causal prefix sum
  scan_chunks<<<dim3(R/256, NCH, B), 256, 0, stream>>>(Kw, Stot, T, R, CH);
  scan_totals<<<dim3(R/256, 1, B), 256, 0, stream>>>(Stot, NCH, R);

  // pass 3a: AS[b,t,r] = (x @ MqvT^T) * (S_local + chunk_off) -> bf16
  gemm_async<EPI_AS_BF16><<<dim3(T/256, R/128, B), 256, 0, stream>>>(
      xbf, MqvT, D, D, D, (long)T*D,
      AS, R, (long)T*R,
      Kw, (long)T*R, Stot, (long)NCH*R);

  // pass 3b: out[b,t,e] = (AS @ N2^T)/max(n,1) + (n>0)*c[e]
  gemm_async<EPI_FINAL><<<dim3(T/256, D/128, B), 256, 0, stream>>>(
      AS, N2, R, R, R, (long)T*R,
      d_out, D, (long)T*D,
      nvec, (long)T, cvec, 0);
}
